// SetTransformer_65309272703516
// MI455X (gfx1250) — matrix operand order repaired and checked by reading, not compiled, not run on hardware
//
#include <hip/hip_runtime.h>
#include <hip/hip_bf16.h>

// ---------------------------------------------------------------------------
// Set Transformer forward for MI455X (gfx1250, wave32, WMMA).
// All matmuls use v_wmma_f32_16x16x32_f16 (f16 in, f32 accumulate).
// Fragment layouts used here (same contiguous K order in A and B):
//   A (16x32 f16): lane l holds row (l&15); element e (j=e>>1,p=e&1) holds
//                  K = 2j + p + 16*(l>>4)
//   B (32x16 f16): lane l holds col (l&15); element e holds K = e + 16*(l>>4)
//   C/D (16x16 f32): lane l holds col (l&15); VGPR r holds row r + 8*(l>>4)
// ---------------------------------------------------------------------------

typedef __attribute__((ext_vector_type(16))) _Float16 v16h;
typedef __attribute__((ext_vector_type(8)))  float    v8f;

#define WMMA_F16(A, B, C) \
    __builtin_amdgcn_wmma_f32_16x16x32_f16(false, (A), false, (B), (short)0, (C), false, false)

// ---------------------------------------------------------------------------
// Weight conversion: W[64,K] f32 -> Wh[64,Kpad] f16, zero-padded in K.
// ---------------------------------------------------------------------------
__global__ void convert_w_kernel(const float* __restrict__ W,
                                 _Float16* __restrict__ Wh, int K, int Kpad) {
    int i = blockIdx.x * 256 + threadIdx.x;
    if (i >= 64 * Kpad) return;
    int n = i / Kpad, k = i % Kpad;
    Wh[i] = (k < K) ? (_Float16)W[n * K + k] : (_Float16)0.0f;
}

// ---------------------------------------------------------------------------
// GEMM: Y[M,64] = X[M,K] @ Wh[64,Kpad]^T (+bias) with optional epilogue
//   mode 0: Y = acc + bias
//   mode 1: Y = resid + relu(acc + bias)
// Block: 256 threads (8 waves), 128 rows x 64 cols per block. M = 65536.
// Requires K % 4 == 0 (true: 200 and 64).
// ---------------------------------------------------------------------------
__global__ __launch_bounds__(256) void gemm_kernel(
    const float* __restrict__ X, const _Float16* __restrict__ Wh,
    const float* __restrict__ bias, const float* __restrict__ resid,
    float* __restrict__ Y, int K, int Kpad, int mode) {
    __shared__ __attribute__((aligned(16))) _Float16 sX[128 * 32];

    const int tid  = threadIdx.x;
    const int wave = tid >> 5, lane = tid & 31;
    const int ln = lane & 15, hi = lane >> 4;
    const int rowBase = blockIdx.x * 128;

    // Staging geometry: thread covers row (tid>>1), 16 K-columns at (tid&1)*16.
    const int sr = tid >> 1;
    const int sk = (tid & 1) * 16;
    const float* srow = X + (size_t)(rowBase + sr) * K;
    uint4* sdst = (uint4*)(sX + sr * 32 + sk);

    v8f acc[4];
#pragma unroll
    for (int g = 0; g < 4; ++g)
#pragma unroll
        for (int i = 0; i < 8; ++i) acc[g][i] = 0.0f;

    for (int kb = 0; kb < Kpad; kb += 32) {
        // Prefetch next K-tile of X into cache (global_prefetch_b8).
        if (kb + 32 < K) __builtin_prefetch(srow + kb + 32 + sk, 0, 0);

        __syncthreads();
        // Stage 128x32 A tile as f16 into LDS via float4 loads + b128 stores.
        {
            union { _Float16 h[16]; uint4 q[2]; } t;
            const float4* src4 = (const float4*)(srow + kb + sk);
            if (kb + 32 <= K) {  // uniform fast path: no guards
#pragma unroll
                for (int jj = 0; jj < 4; ++jj) {
                    float4 f = src4[jj];
                    t.h[4 * jj + 0] = (_Float16)f.x;
                    t.h[4 * jj + 1] = (_Float16)f.y;
                    t.h[4 * jj + 2] = (_Float16)f.z;
                    t.h[4 * jj + 3] = (_Float16)f.w;
                }
            } else {             // tail (last step of K=200 only), float4 guard
#pragma unroll
                for (int jj = 0; jj < 4; ++jj) {
                    if (kb + sk + jj * 4 + 4 <= K) {
                        float4 f = src4[jj];
                        t.h[4 * jj + 0] = (_Float16)f.x;
                        t.h[4 * jj + 1] = (_Float16)f.y;
                        t.h[4 * jj + 2] = (_Float16)f.z;
                        t.h[4 * jj + 3] = (_Float16)f.w;
                    } else {
                        t.h[4 * jj + 0] = (_Float16)0.0f;
                        t.h[4 * jj + 1] = (_Float16)0.0f;
                        t.h[4 * jj + 2] = (_Float16)0.0f;
                        t.h[4 * jj + 3] = (_Float16)0.0f;
                    }
                }
            }
            sdst[0] = t.q[0];
            sdst[1] = t.q[1];
        }
        __syncthreads();

        // Build A fragment for this wave's 16-row strip.
        union { v16h v; unsigned u[8]; } A;
        const unsigned* sXu = (const unsigned*)(sX + (wave * 16 + ln) * 32);
#pragma unroll
        for (int j = 0; j < 8; ++j) {
            int kk = 2 * j + 16 * hi;  // even
            A.u[j] = sXu[kk >> 1];
        }

        // 4 column groups of 16: B fragments read straight from f16 weights.
#pragma unroll
        for (int g = 0; g < 4; ++g) {
            const _Float16* wp = Wh + (size_t)(g * 16 + ln) * Kpad + kb + 16 * hi;
            union { v16h v; uint4 q[2]; } B;
            B.q[0] = *(const uint4*)(wp);
            B.q[1] = *(const uint4*)(wp + 8);
            acc[g] = WMMA_F16(A.v, B.v, acc[g]);
        }
    }

    // Epilogue.
#pragma unroll
    for (int g = 0; g < 4; ++g) {
        int n = g * 16 + ln;
        float bv = bias[n];
#pragma unroll
        for (int r = 0; r < 8; ++r) {
            int m = rowBase + wave * 16 + r + 8 * hi;
            size_t idx = (size_t)m * 64 + n;
            float val = acc[g][r] + bv;
            if (mode) val = resid[idx] + fmaxf(val, 0.0f);
            Y[idx] = val;
        }
    }
}

// ---------------------------------------------------------------------------
// Fused masked attention for one (batch, head): one workgroup of 16 waves,
// each wave owns 16 query rows. Scores live in 16 WMMA accumulators
// (16x256 per wave); masked softmax via xor-shuffles in 16-lane halves;
// P re-laid-out through a per-wave LDS slice into A fragments for P@V.
// O[b,m,h*16+d] = Qp[b,m,h*16+d] + sum_key P * V.
// ---------------------------------------------------------------------------
__global__ __launch_bounds__(512) void attn_kernel(
    const float* __restrict__ Qp, const float* __restrict__ Kp,
    const float* __restrict__ Vp, const int* __restrict__ lengths,
    float* __restrict__ O) {
    __shared__ __attribute__((aligned(16))) _Float16 sK[256 * 16];   // [key][d]
    __shared__ __attribute__((aligned(16))) _Float16 sVT[16 * 256];  // [d][key]
    __shared__ __attribute__((aligned(16))) _Float16 sP[16][16 * 32];// per-wave

    const int b = blockIdx.x >> 2, h = blockIdx.x & 3;
    const int len = lengths[b];
    const int tid = threadIdx.x;
    const int wave = tid >> 5, lane = tid & 31;
    const int ln = lane & 15, hi = lane >> 4;
    const float scale = 0.125f;  // 1/sqrt(64)

    // Stage K (row-major) and V^T (f16) for this (b,h), float4-vectorized:
    // i indexes 1024 float4 chunks: key m = i>>2, d-quad dq = i&3.
    const float* Kb = Kp + ((size_t)b * 256) * 64 + h * 16;
    const float* Vb = Vp + ((size_t)b * 256) * 64 + h * 16;
    for (int i = tid; i < 1024; i += 512) {
        int m = i >> 2, dq = (i & 3) * 4;
        float4 fk = *(const float4*)(Kb + (size_t)m * 64 + dq);
        float4 fv = *(const float4*)(Vb + (size_t)m * 64 + dq);
        union { _Float16 h[4]; unsigned u2[2]; } tk;
        tk.h[0] = (_Float16)fk.x; tk.h[1] = (_Float16)fk.y;
        tk.h[2] = (_Float16)fk.z; tk.h[3] = (_Float16)fk.w;
        *(unsigned*)(sK + m * 16 + dq) = tk.u2[0];
        *(unsigned*)(sK + m * 16 + dq + 2) = tk.u2[1];
        sVT[(dq + 0) * 256 + m] = (_Float16)fv.x;
        sVT[(dq + 1) * 256 + m] = (_Float16)fv.y;
        sVT[(dq + 2) * 256 + m] = (_Float16)fv.z;
        sVT[(dq + 3) * 256 + m] = (_Float16)fv.w;
    }
    __syncthreads();

    // Q A-fragment: rows qBase..qBase+15, K-dim 16 zero-padded to 32.
    const int qBase = wave * 16;
    const float* Qb = Qp + ((size_t)b * 256 + qBase) * 64 + h * 16;
    union { v16h v; _Float16 e[16]; } Aq;
#pragma unroll
    for (int e = 0; e < 16; ++e) {
        int j = e >> 1, p = e & 1;
        int k = e + 16 * hi;
        Aq.e[e] = (k < 16) ? (_Float16)Qb[(size_t)ln * 64 + k] : (_Float16)0.0f;
    }

    // 16 score tiles: S = Q @ K^T (K-dim padded; hi lanes carry zeros).
    v8f S[16];
#pragma unroll
    for (int j = 0; j < 16; ++j) {
        union { v16h v; uint4 q[2]; } Bk;
        if (hi == 0) {
            const _Float16* kp = sK + (j * 16 + ln) * 16;
            Bk.q[0] = *(const uint4*)kp;
            Bk.q[1] = *(const uint4*)(kp + 8);
        } else {
            uint4 z = {0u, 0u, 0u, 0u};
            Bk.q[0] = z; Bk.q[1] = z;
        }
        v8f zc;
#pragma unroll
        for (int i = 0; i < 8; ++i) zc[i] = 0.0f;
        S[j] = WMMA_F16(Aq.v, Bk.v, zc);
    }

    // Apply mask + scale in place (once).
#pragma unroll
    for (int j = 0; j < 16; ++j) {
        bool valid = (j * 16 + ln) < len;
#pragma unroll
        for (int r = 0; r < 8; ++r)
            S[j][r] = valid ? S[j][r] * scale : -1e9f;
    }

    // Softmax per accumulator row r (row = qBase + r + 8*hi):
    // row-max via xor-shuffles in 16-lane halves, then overwrite S with
    // exp(S - max), then row-sum.
    float inv[8];
#pragma unroll
    for (int r = 0; r < 8; ++r) {
        float mx = -1e30f;
#pragma unroll
        for (int j = 0; j < 16; ++j) mx = fmaxf(mx, S[j][r]);
        for (int d = 1; d < 16; d <<= 1) mx = fmaxf(mx, __shfl_xor(mx, d, 32));
        float sum = 0.0f;
#pragma unroll
        for (int j = 0; j < 16; ++j) {
            float e = __expf(S[j][r] - mx);
            S[j][r] = e;
            sum += e;
        }
        for (int d = 1; d < 16; d <<= 1) sum += __shfl_xor(sum, d, 32);
        inv[r] = 1.0f / sum;
    }

    // O = P @ V over 8 key-chunks of 32. P relaid through per-wave LDS slice
    // (same-wave DS ops are in-order, no block barrier needed).
    v8f Oacc;
#pragma unroll
    for (int i = 0; i < 8; ++i) Oacc[i] = 0.0f;
    _Float16* myP = &sP[wave][0];

    for (int c = 0; c < 8; ++c) {
#pragma unroll
        for (int jj = 0; jj < 2; ++jj) {
            int j = c * 2 + jj;
#pragma unroll
            for (int r = 0; r < 8; ++r) {
                float p = S[j][r] * inv[r];
                myP[(r + 8 * hi) * 32 + jj * 16 + ln] = (_Float16)p;
            }
        }
        // A fragment (P rows) from own LDS slice.
        union { v16h v; unsigned u[8]; } Ap;
        const unsigned* pu = (const unsigned*)(myP + ln * 32);
#pragma unroll
        for (int j2 = 0; j2 < 8; ++j2) {
            int kk = 2 * j2 + 16 * hi;
            Ap.u[j2] = pu[kk >> 1];
        }
        // B fragment (V^T): contiguous 16 keys per lane.
        union { v16h v; uint4 q[2]; } Bv;
        const _Float16* vp = sVT + ln * 256 + c * 32 + 16 * hi;
        Bv.q[0] = *(const uint4*)vp;
        Bv.q[1] = *(const uint4*)(vp + 8);
        Oacc = WMMA_F16(Ap.v, Bv.v, Oacc);
    }

    // Residual + store.
    const float* Qrow = Qp + ((size_t)b * 256) * 64 + h * 16;
    float* Orow = O + ((size_t)b * 256) * 64 + h * 16;
#pragma unroll
    for (int r = 0; r < 8; ++r) {
        int m = qBase + r + 8 * hi;
        size_t idx = (size_t)m * 64 + ln;
        Orow[idx] = Qrow[idx] + Oacc[r];
    }
}

// ---------------------------------------------------------------------------
// PMA (Nq=1) + output head. One block of 256 threads per batch.
// Kp3/Vp3 precomputed by gemm_kernel. Tiny FLOPs -> scalar VALU is fine.
// ---------------------------------------------------------------------------
__global__ __launch_bounds__(256) void pma_kernel(
    const float* __restrict__ Kp, const float* __restrict__ Vp,
    const int* __restrict__ lengths,
    const float* __restrict__ wq3, const float* __restrict__ bq3,
    const float* __restrict__ wo3, const float* __restrict__ bo3,
    const float* __restrict__ seed, const float* __restrict__ wfc,
    const float* __restrict__ bfc, float* __restrict__ out) {
    __shared__ float qp[64], oh[64], Ovec[64], emb[64];
    __shared__ float red[256];
    const int b = blockIdx.x, tid = threadIdx.x;
    const int len = lengths[b];

    if (tid < 64) {
        float a = bq3[tid];
        for (int k = 0; k < 64; ++k) a += seed[k] * wq3[tid * 64 + k];
        qp[tid] = a;
        oh[tid] = 0.0f;
    }
    __syncthreads();

    const float* Kb = Kp + ((size_t)b * 256 + tid) * 64;
    const float* Vb = Vp + ((size_t)b * 256 + tid) * 64;
    float a_h[4];
    for (int h = 0; h < 4; ++h) {
        float s = 0.0f;
        for (int d = 0; d < 16; ++d) s += qp[h * 16 + d] * Kb[h * 16 + d];
        s *= 0.125f;
        s = (tid < len) ? s : -1e9f;
        red[tid] = s; __syncthreads();
        for (int off = 128; off > 0; off >>= 1) {
            if (tid < off) red[tid] = fmaxf(red[tid], red[tid + off]);
            __syncthreads();
        }
        float mx = red[0]; __syncthreads();
        float e = __expf(s - mx);
        red[tid] = e; __syncthreads();
        for (int off = 128; off > 0; off >>= 1) {
            if (tid < off) red[tid] += red[tid + off];
            __syncthreads();
        }
        a_h[h] = e / red[0];
        __syncthreads();
    }
    for (int h = 0; h < 4; ++h)
        for (int d = 0; d < 16; ++d)
            atomicAdd(&oh[h * 16 + d], a_h[h] * Vb[h * 16 + d]);
    __syncthreads();
    if (tid < 64) Ovec[tid] = qp[tid] + oh[tid];
    __syncthreads();
    if (tid < 64) {
        float a = bo3[tid];
        for (int k = 0; k < 64; ++k) a += Ovec[k] * wo3[tid * 64 + k];
        float e = Ovec[tid] + fmaxf(a, 0.0f);
        emb[tid] = e;
        out[256 + (size_t)b * 64 + tid] = e;   // emb output
    }
    __syncthreads();
    if (tid == 0) {
        float a = bfc[0];
        for (int k = 0; k < 64; ++k) a += emb[k] * wfc[k];
        out[b] = 1.0f / (1.0f + __expf(-a));   // sigmoid output
    }
}

// ---------------------------------------------------------------------------
extern "C" void kernel_launch(void* const* d_in, const int* in_sizes, int n_in,
                              void* d_out, int out_size, void* d_ws, size_t ws_size,
                              hipStream_t stream) {
    (void)in_sizes; (void)n_in; (void)out_size; (void)ws_size;
    const float* x = (const float*)d_in[0];
    const int* lengths = (const int*)d_in[1];
    const float *bq1 = (const float*)d_in[3];
    const float *bk1 = (const float*)d_in[5];
    const float *bv1 = (const float*)d_in[7];
    const float *wo1 = (const float*)d_in[8],  *bo1 = (const float*)d_in[9];
    const float *wq2 = (const float*)d_in[10], *bq2 = (const float*)d_in[11];
    const float *wk2 = (const float*)d_in[12], *bk2 = (const float*)d_in[13];
    const float *wv2 = (const float*)d_in[14], *bv2 = (const float*)d_in[15];
    const float *wo2 = (const float*)d_in[16], *bo2 = (const float*)d_in[17];
    const float *wq3 = (const float*)d_in[18], *bq3 = (const float*)d_in[19];
    const float *wk3 = (const float*)d_in[20], *bk3 = (const float*)d_in[21];
    const float *wv3 = (const float*)d_in[22], *bv3 = (const float*)d_in[23];
    const float *wo3 = (const float*)d_in[24], *bo3 = (const float*)d_in[25];
    const float *seed = (const float*)d_in[26];
    const float *wfc = (const float*)d_in[27], *bfc = (const float*)d_in[28];
    float* out = (float*)d_out;

    // Workspace layout: f16 weights, then 5 f32 buffers of [65536,64].
    char* ws = (char*)d_ws;
    size_t off = 0;
    auto allocH = [&](size_t halves) {
        _Float16* p = (_Float16*)(ws + off); off += halves * sizeof(_Float16); return p;
    };
    _Float16* wq1h = allocH(64 * 224);
    _Float16* wk1h = allocH(64 * 224);
    _Float16* wv1h = allocH(64 * 224);
    _Float16* wo1h = allocH(64 * 64);
    _Float16* wq2h = allocH(64 * 64);
    _Float16* wk2h = allocH(64 * 64);
    _Float16* wv2h = allocH(64 * 64);
    _Float16* wo2h = allocH(64 * 64);
    _Float16* wk3h = allocH(64 * 64);
    _Float16* wv3h = allocH(64 * 64);
    off = (off + 255) & ~(size_t)255;
    const size_t BUF = (size_t)65536 * 64 * sizeof(float);
    float* bufQ = (float*)(ws + off); off += BUF;
    float* bufK = (float*)(ws + off); off += BUF;
    float* bufV = (float*)(ws + off); off += BUF;
    float* bufO = (float*)(ws + off); off += BUF;
    float* bufH = (float*)(ws + off); off += BUF;

    // Weight conversion.
    const int g224 = (64 * 224 + 255) / 256, g64 = (64 * 64 + 255) / 256;
    convert_w_kernel<<<g224, 256, 0, stream>>>((const float*)d_in[2], wq1h, 200, 224);
    convert_w_kernel<<<g224, 256, 0, stream>>>((const float*)d_in[4], wk1h, 200, 224);
    convert_w_kernel<<<g224, 256, 0, stream>>>((const float*)d_in[6], wv1h, 200, 224);
    convert_w_kernel<<<g64, 256, 0, stream>>>(wo1, wo1h, 64, 64);
    convert_w_kernel<<<g64, 256, 0, stream>>>(wq2, wq2h, 64, 64);
    convert_w_kernel<<<g64, 256, 0, stream>>>(wk2, wk2h, 64, 64);
    convert_w_kernel<<<g64, 256, 0, stream>>>(wv2, wv2h, 64, 64);
    convert_w_kernel<<<g64, 256, 0, stream>>>(wo2, wo2h, 64, 64);
    convert_w_kernel<<<g64, 256, 0, stream>>>(wk3, wk3h, 64, 64);
    convert_w_kernel<<<g64, 256, 0, stream>>>(wv3, wv3h, 64, 64);

    // SAB1.
    gemm_kernel<<<512, 256, 0, stream>>>(x, wq1h, bq1, nullptr, bufQ, 200, 224, 0);
    gemm_kernel<<<512, 256, 0, stream>>>(x, wk1h, bk1, nullptr, bufK, 200, 224, 0);
    gemm_kernel<<<512, 256, 0, stream>>>(x, wv1h, bv1, nullptr, bufV, 200, 224, 0);
    attn_kernel<<<1024, 512, 0, stream>>>(bufQ, bufK, bufV, lengths, bufO);
    gemm_kernel<<<512, 256, 0, stream>>>(bufO, wo1h, bo1, bufO, bufH, 64, 64, 1);

    // SAB2.
    gemm_kernel<<<512, 256, 0, stream>>>(bufH, wq2h, bq2, nullptr, bufQ, 64, 64, 0);
    gemm_kernel<<<512, 256, 0, stream>>>(bufH, wk2h, bk2, nullptr, bufK, 64, 64, 0);
    gemm_kernel<<<512, 256, 0, stream>>>(bufH, wv2h, bv2, nullptr, bufV, 64, 64, 0);
    attn_kernel<<<1024, 512, 0, stream>>>(bufQ, bufK, bufV, lengths, bufO);
    gemm_kernel<<<512, 256, 0, stream>>>(bufO, wo2h, bo2, bufO, bufH, 64, 64, 1);

    // PMA projections (K/V over h2) + pooled head.
    gemm_kernel<<<512, 256, 0, stream>>>(bufH, wk3h, bk3, nullptr, bufK, 64, 64, 0);
    gemm_kernel<<<512, 256, 0, stream>>>(bufH, wv3h, bv3, nullptr, bufV, 64, 64, 0);
    pma_kernel<<<256, 256, 0, stream>>>(bufK, bufV, lengths, wq3, bq3, wo3, bo3,
                                        seed, wfc, bfc, out);
}
